// TrellisExpert_678604833231
// MI455X (gfx1250) — matrix operand order repaired and checked by reading, not compiled, not run on hardware
//
#include <hip/hip_runtime.h>

#define HIDDEN 2048
#define INTER  5632
#define NCODES 4096
#define VEC    8
#define TOKENS 4096

typedef __attribute__((ext_vector_type(16))) __bf16 v16bf;
typedef __attribute__((ext_vector_type(8)))  __bf16 v8bf;
typedef __attribute__((ext_vector_type(4)))  __bf16 v4bf;
typedef __attribute__((ext_vector_type(8)))  float  v8f;
typedef __attribute__((ext_vector_type(4)))  float  v4f;

typedef unsigned int u32x4 __attribute__((ext_vector_type(4)));
typedef int          i32x8 __attribute__((ext_vector_type(8)));
typedef int          i32x4 __attribute__((ext_vector_type(4)));

// LDS tile geometry: K-chunk of 64 bf16 per row = 128B = 32 DWORDs.
// TDM pads 4 DWORDs (16B) after every 32 DWORDs -> LDS row stride 144B.
#define KCHUNK 64
#define LSTR   72   // elements (144 bytes)

// ---------------------------------------------------------------------------
// Dequant: w[r, c*8 + j] = codebook[idx[r,c]][j] * scale[r], stored as bf16.
// ---------------------------------------------------------------------------
__global__ void __launch_bounds__(256)
dequant_bf16_kernel(const float* __restrict__ codebook,
                    const int*   __restrict__ idx,
                    const float* __restrict__ scale,
                    __bf16*      __restrict__ out,
                    int rows, int cgroups) {
    int tid = blockIdx.x * blockDim.x + threadIdx.x;
    int total = rows * cgroups;
    if (tid >= total) return;
    int r = tid / cgroups;
    int code = idx[tid];
    const v4f* cb = (const v4f*)(codebook + (size_t)code * VEC);
    v4f lo = cb[0];
    v4f hi = cb[1];
    float s = scale[r];
    v8bf o;
    o[0] = (__bf16)(lo[0] * s); o[1] = (__bf16)(lo[1] * s);
    o[2] = (__bf16)(lo[2] * s); o[3] = (__bf16)(lo[3] * s);
    o[4] = (__bf16)(hi[0] * s); o[5] = (__bf16)(hi[1] * s);
    o[6] = (__bf16)(hi[2] * s); o[7] = (__bf16)(hi[3] * s);
    *(v8bf*)(out + (size_t)tid * VEC) = o;
}

// ---------------------------------------------------------------------------
// fp32 -> bf16 conversion of x (4 elements per thread).
// ---------------------------------------------------------------------------
__global__ void __launch_bounds__(256)
f32_to_bf16_kernel(const float* __restrict__ in, __bf16* __restrict__ out, int n4) {
    int t = blockIdx.x * blockDim.x + threadIdx.x;
    if (t >= n4) return;
    v4f v = ((const v4f*)in)[t];
    v4bf o;
    o[0] = (__bf16)v[0]; o[1] = (__bf16)v[1];
    o[2] = (__bf16)v[2]; o[3] = (__bf16)v[3];
    ((v4bf*)out)[t] = o;
}

// ---------------------------------------------------------------------------
// TDM: async 2-D tile load Global -> LDS (rows x 64 bf16, padded to 144B rows).
// Descriptor packing per CDNA5 ISA.
// 6-arg builtin form (clang-23 / therock-10.0 headers).
// ---------------------------------------------------------------------------
__device__ __forceinline__ void tdm_load_2d(unsigned lds_byte_off,
                                            const __bf16* gsrc,
                                            int rows, int stride_elems) {
    unsigned long long ga = (unsigned long long)(uintptr_t)gsrc;
    u32x4 g0;
    g0.x = 1u;                                           // count=1 (user mode)
    g0.y = lds_byte_off;                                 // lds_addr
    g0.z = (unsigned)ga;                                 // global_addr[31:0]
    g0.w = (unsigned)((ga >> 32) & 0x01FFFFFFu)          // global_addr[56:32]
         | (2u << 30);                                   // type = 2 ("image")
    i32x8 g1;
    g1[0] = (1 << 16)        // data_size = 2 bytes
          | (1 << 20)        // pad_enable
          | (4 << 22)        // pad_interval: 32 DWORDs (= one 64-elem row)
          | (3 << 25);       // pad_amount: 4 DWORDs (16B) -> 144B LDS stride
    g1[1] = (stride_elems & 0xFFFF) << 16;               // tensor_dim0[15:0]
    g1[2] = ((stride_elems >> 16) & 0xFFFF)              // tensor_dim0[31:16]
          | ((rows & 0xFFFF) << 16);                     // tensor_dim1[15:0]
    g1[3] = (KCHUNK & 0xFFFF) << 16;                     // dim1 hi=0 | tile_dim0=64
    g1[4] = rows & 0xFFFF;                               // tile_dim1 | tile_dim2=0
    g1[5] = stride_elems;                                // tensor_dim0_stride[31:0]
    g1[6] = 0;
    g1[7] = 0;
    i32x4 z4 = {0, 0, 0, 0};                             // groups 2/3: 2-D tensor
    i32x8 z8 = {0, 0, 0, 0, 0, 0, 0, 0};
    __builtin_amdgcn_tensor_load_to_lds(g0, g1, z4, z4, z8, 0);
}

// ---------------------------------------------------------------------------
// WMMA fragment loaders (wave32 v_wmma_f32_16x16x32_bf16 layouts per ISA):
// A 16x32: lane row = lane%16; K chunks [kg,kg+8) and [16+kg,16+kg+8), kg=(lane/16)*8
// B 32x16: lane col = lane%16; K chunks [kg,kg+8) and [16+kg,16+kg+8), kg=(lane/16)*8
// ---------------------------------------------------------------------------
__device__ __forceinline__ v16bf load_a_frag(const __bf16* rowk, int lane) {
    int kg = (lane >> 4) << 3;
    union { v16bf v; v8bf h[2]; } u;
    u.h[0] = *(const v8bf*)(rowk + kg);
    u.h[1] = *(const v8bf*)(rowk + 16 + kg);
    return u.v;
}

__device__ __forceinline__ v16bf load_b_frag(const __bf16* rowk, int lane) {
    int kb = (lane >> 4) << 3;
    union { v16bf v; v8bf h[2]; } u;
    u.h[0] = *(const v8bf*)(rowk + kb);
    u.h[1] = *(const v8bf*)(rowk + 16 + kb);
    return u.v;
}

// ---------------------------------------------------------------------------
// GEMM1 fused SwiGLU: h = silu(x @ Wg^T) * (x @ Wu^T), bf16 out.
// Block tile 64(M) x 128(N); 8 waves as 4(M) x 2(N); wave tile 16x64.
// K-chunks of 64 staged in LDS by TDM, double buffered.
// Inner loop: batch all 9 fragment loads (18 ds_load_b128), then 8 WMMAs.
// LDS/buffer: A 64x72 + Bg 128x72 + Bu 128x72 = 23040 elems (45KB); x2 = 90KB.
// ---------------------------------------------------------------------------
__global__ void __launch_bounds__(256)
gemm1_swiglu_kernel(const __bf16* __restrict__ x,
                    const __bf16* __restrict__ wg,
                    const __bf16* __restrict__ wu,
                    __bf16*       __restrict__ h) {
    extern __shared__ __bf16 smem[];
    const int lane = threadIdx.x & 31;
    const int wave = threadIdx.x >> 5;
    const int NBN  = INTER / 128;               // 44
    const int bm   = blockIdx.x / NBN;          // 0..63
    const int bn   = blockIdx.x - bm * NBN;
    const int mW   = wave >> 1;                 // 0..3
    const int nW   = wave & 1;                  // 0..1

    const int A_ELEMS = 64 * LSTR;
    const int B_ELEMS = 128 * LSTR;
    const int BUF     = A_ELEMS + 2 * B_ELEMS;  // 23040 elems
    const int NC      = HIDDEN / KCHUNK;        // 32 chunks

    const __bf16* gA = x  + (size_t)(bm * 64)  * HIDDEN;
    const __bf16* gG = wg + (size_t)(bn * 128) * HIDDEN;
    const __bf16* gU = wu + (size_t)(bn * 128) * HIDDEN;

    v8f accg[4] = {v8f{}, v8f{}, v8f{}, v8f{}};
    v8f accu[4] = {v8f{}, v8f{}, v8f{}, v8f{}};

    if (wave == 0) {
        unsigned base = (unsigned)(uintptr_t)smem;
        tdm_load_2d(base,                            gA, 64,  HIDDEN);
        tdm_load_2d(base + A_ELEMS * 2,              gG, 128, HIDDEN);
        tdm_load_2d(base + (A_ELEMS + B_ELEMS) * 2,  gU, 128, HIDDEN);
    }

    for (int c = 0; c < NC; ++c) {
        if (wave == 0) __builtin_amdgcn_s_wait_tensorcnt(0);
        __syncthreads();                              // buf[c&1] ready
        if (wave == 0 && c + 1 < NC) {
            unsigned base = (unsigned)(uintptr_t)(smem + ((c + 1) & 1) * BUF);
            int ko = (c + 1) * KCHUNK;
            tdm_load_2d(base,                           gA + ko, 64,  HIDDEN);
            tdm_load_2d(base + A_ELEMS * 2,             gG + ko, 128, HIDDEN);
            tdm_load_2d(base + (A_ELEMS + B_ELEMS) * 2, gU + ko, 128, HIDDEN);
        }
        const __bf16* Ab = smem + (c & 1) * BUF;
        const __bf16* Gb = Ab + A_ELEMS;
        const __bf16* Ub = Gb + B_ELEMS;
        const __bf16* aRow = Ab + (mW * 16 + (lane & 15)) * LSTR;
#pragma unroll
        for (int s = 0; s < 2; ++s) {                 // two k-steps of 32
            // Phase 1: batch all fragment loads for this k-step.
            v16bf a = load_a_frag(aRow + s * 32, lane);
            v16bf bg[4], bu[4];
#pragma unroll
            for (int t = 0; t < 4; ++t) {
                int brow = (nW * 64 + t * 16 + (lane & 15)) * LSTR + s * 32;
                bg[t] = load_b_frag(Gb + brow, lane);
                bu[t] = load_b_frag(Ub + brow, lane);
            }
            // Phase 2: back-to-back WMMAs.
#pragma unroll
            for (int t = 0; t < 4; ++t) {
                accg[t] = __builtin_amdgcn_wmma_f32_16x16x32_bf16(
                    false, a, false, bg[t], (short)0, accg[t], false, false);
                accu[t] = __builtin_amdgcn_wmma_f32_16x16x32_bf16(
                    false, a, false, bu[t], (short)0, accu[t], false, false);
            }
        }
        __syncthreads();                              // done reading buf[c&1]
    }

    // Epilogue: D element (M = v + 8*(lane/16), N = lane%16); silu(g)*u -> bf16
    const int rowBase = bm * 64 + mW * 16 + ((lane >> 4) << 3);
#pragma unroll
    for (int t = 0; t < 4; ++t) {
        int col = bn * 128 + nW * 64 + t * 16 + (lane & 15);
#pragma unroll
        for (int v = 0; v < 8; ++v) {
            float g   = accg[t][v];
            float u   = accu[t][v];
            float sig = 1.0f / (1.0f + __expf(-g));
            h[(size_t)(rowBase + v) * INTER + col] = (__bf16)(g * sig * u);
        }
    }
}

// ---------------------------------------------------------------------------
// GEMM2: out = h @ Wd^T, fp32 out.
// Block tile 128(M) x 128(N); 8 waves as 4(M) x 2(N); wave tile 32x64
// (2 M-tiles x 4 N-tiles = 8 accumulators; batched loads then 8 WMMAs).
// LDS/buffer: A 128x72 + B 128x72 = 18432 elems (36KB); x2 = 72KB.
// ---------------------------------------------------------------------------
__global__ void __launch_bounds__(256)
gemm2_kernel(const __bf16* __restrict__ h,
             const __bf16* __restrict__ wd,
             float*        __restrict__ out) {
    extern __shared__ __bf16 smem[];
    const int lane = threadIdx.x & 31;
    const int wave = threadIdx.x >> 5;
    const int NBN  = HIDDEN / 128;              // 16
    const int bm   = blockIdx.x / NBN;          // 0..31
    const int bn   = blockIdx.x - bm * NBN;
    const int mW   = wave >> 1;                 // 0..3
    const int nW   = wave & 1;                  // 0..1

    const int T_ELEMS = 128 * LSTR;
    const int BUF     = 2 * T_ELEMS;            // 18432 elems
    const int NC      = INTER / KCHUNK;         // 88 chunks

    const __bf16* gA = h  + (size_t)(bm * 128) * INTER;
    const __bf16* gB = wd + (size_t)(bn * 128) * INTER;

    v8f acc[2][4] = {{v8f{}, v8f{}, v8f{}, v8f{}},
                     {v8f{}, v8f{}, v8f{}, v8f{}}};

    if (wave == 0) {
        unsigned base = (unsigned)(uintptr_t)smem;
        tdm_load_2d(base,               gA, 128, INTER);
        tdm_load_2d(base + T_ELEMS * 2, gB, 128, INTER);
    }

    for (int c = 0; c < NC; ++c) {
        if (wave == 0) __builtin_amdgcn_s_wait_tensorcnt(0);
        __syncthreads();
        if (wave == 0 && c + 1 < NC) {
            unsigned base = (unsigned)(uintptr_t)(smem + ((c + 1) & 1) * BUF);
            int ko = (c + 1) * KCHUNK;
            tdm_load_2d(base,               gA + ko, 128, INTER);
            tdm_load_2d(base + T_ELEMS * 2, gB + ko, 128, INTER);
        }
        const __bf16* Ab = smem + (c & 1) * BUF;
        const __bf16* Bb = Ab + T_ELEMS;
#pragma unroll
        for (int s = 0; s < 2; ++s) {
            // Phase 1: batch all fragment loads for this k-step.
            v16bf a0 = load_a_frag(Ab + (mW * 32 + (lane & 15)) * LSTR + s * 32, lane);
            v16bf a1 = load_a_frag(Ab + (mW * 32 + 16 + (lane & 15)) * LSTR + s * 32, lane);
            v16bf b[4];
#pragma unroll
            for (int t = 0; t < 4; ++t) {
                int brow = (nW * 64 + t * 16 + (lane & 15)) * LSTR + s * 32;
                b[t] = load_b_frag(Bb + brow, lane);
            }
            // Phase 2: back-to-back WMMAs.
#pragma unroll
            for (int t = 0; t < 4; ++t) {
                acc[0][t] = __builtin_amdgcn_wmma_f32_16x16x32_bf16(
                    false, a0, false, b[t], (short)0, acc[0][t], false, false);
                acc[1][t] = __builtin_amdgcn_wmma_f32_16x16x32_bf16(
                    false, a1, false, b[t], (short)0, acc[1][t], false, false);
            }
        }
        __syncthreads();
    }

#pragma unroll
    for (int mt = 0; mt < 2; ++mt) {
        const int rowBase = bm * 128 + mW * 32 + mt * 16 + ((lane >> 4) << 3);
#pragma unroll
        for (int t = 0; t < 4; ++t) {
            int col = bn * 128 + nW * 64 + t * 16 + (lane & 15);
#pragma unroll
            for (int v = 0; v < 8; ++v) {
                out[(size_t)(rowBase + v) * HIDDEN + col] = acc[mt][t][v];
            }
        }
    }
}

// ---------------------------------------------------------------------------
// Launch
// ---------------------------------------------------------------------------
extern "C" void kernel_launch(void* const* d_in, const int* in_sizes, int n_in,
                              void* d_out, int out_size, void* d_ws, size_t ws_size,
                              hipStream_t stream) {
    const float* x      = (const float*)d_in[0];
    const float* cb_g   = (const float*)d_in[1];
    const int*   idx_g  = (const int*)  d_in[2];
    const float* sc_g   = (const float*)d_in[3];
    const float* cb_u   = (const float*)d_in[4];
    const int*   idx_u  = (const int*)  d_in[5];
    const float* sc_u   = (const float*)d_in[6];
    const float* cb_d   = (const float*)d_in[7];
    const int*   idx_d  = (const int*)  d_in[8];
    const float* sc_d   = (const float*)d_in[9];
    float* out = (float*)d_out;

    // Workspace (bf16): Wg | Wu | Wd | x_bf | h_bf  (~132 MB, L2-resident)
    __bf16* ws = (__bf16*)d_ws;
    __bf16* wg = ws;
    __bf16* wu = wg + (size_t)INTER * HIDDEN;
    __bf16* wd = wu + (size_t)INTER * HIDDEN;
    __bf16* xb = wd + (size_t)HIDDEN * INTER;
    __bf16* hb = xb + (size_t)TOKENS * HIDDEN;

    // 1) Dequantize all three weight matrices to bf16.
    {
        int total  = INTER * (HIDDEN / VEC);
        int blocks = (total + 255) / 256;
        dequant_bf16_kernel<<<blocks, 256, 0, stream>>>(cb_g, idx_g, sc_g, wg,
                                                        INTER, HIDDEN / VEC);
        dequant_bf16_kernel<<<blocks, 256, 0, stream>>>(cb_u, idx_u, sc_u, wu,
                                                        INTER, HIDDEN / VEC);
    }
    {
        int total  = HIDDEN * (INTER / VEC);
        int blocks = (total + 255) / 256;
        dequant_bf16_kernel<<<blocks, 256, 0, stream>>>(cb_d, idx_d, sc_d, wd,
                                                        HIDDEN, INTER / VEC);
    }

    // 2) x fp32 -> bf16.
    {
        int n4 = (TOKENS * HIDDEN) / 4;
        int blocks = (n4 + 255) / 256;
        f32_to_bf16_kernel<<<blocks, 256, 0, stream>>>(x, xb, n4);
    }

    // 3) Fused gate/up GEMM + SwiGLU.  Blocks: (4096/64)*(5632/128) = 2816.
    {
        int blocks = (TOKENS / 64) * (INTER / 128);
        size_t lds = (size_t)2 * (64 + 128 + 128) * LSTR * sizeof(__bf16); // 92160B
        gemm1_swiglu_kernel<<<blocks, 256, lds, stream>>>(xb, wg, wu, hb);
    }

    // 4) Down GEMM.  Blocks: (4096/128)*(2048/128) = 512.
    {
        int blocks = (TOKENS / 128) * (HIDDEN / 128);
        size_t lds = (size_t)2 * (128 + 128) * LSTR * sizeof(__bf16);      // 73728B
        gemm2_kernel<<<blocks, 256, lds, stream>>>(hb, wd, out);
    }
}
